// Attention_69114613730156
// MI455X (gfx1250) — matrix operand order repaired and checked by reading, not compiled, not run on hardware
//
#include <hip/hip_runtime.h>

typedef __attribute__((ext_vector_type(16))) __bf16 v16bf;
typedef __attribute__((ext_vector_type(8)))  float  v8f;

#define EMBED 1024
#define NTOK  2048
#define NBATCH 2
#define NHEAD 16
#define HD    64
#define ATT_SCALE 0.125f   // 64^-0.5

// ---------------- CDNA5 async copy helpers ----------------

// GLOBAL_LOAD_ASYNC_TO_LDS_B128 (GV mode): per-lane LDS[vdst] = MEM[vaddr], 16B.
// Tracked with ASYNCcnt. Generic LDS pointer low 32 bits == LDS byte address.
__device__ __forceinline__ void async_load_b128(const __bf16* gaddr, __bf16* lds) {
    uint32_t l = (uint32_t)(uintptr_t)lds;
    uint64_t g = (uint64_t)(uintptr_t)gaddr;
    asm volatile("global_load_async_to_lds_b128 %0, %1, off"
                 :: "v"(l), "v"(g) : "memory");
}

__device__ __forceinline__ void wait_async0() {
#if __has_builtin(__builtin_amdgcn_s_wait_asynccnt)
    __builtin_amdgcn_s_wait_asynccnt(0);
#else
    asm volatile("s_wait_asynccnt 0" ::: "memory");
#endif
}

// ---------------- WMMA fragment helpers per CDNA5 ISA layouts ----------------

// A-matrix (16x32 bf16): lane m=lane&15 holds row m; hi=lane>>4 selects K sub-block.
__device__ __forceinline__ v16bf load_a_frag(const __bf16* rowp, int hi) {
    v16bf a;
#pragma unroll
    for (int e = 0; e < 8; ++e) {
        a[e]     = rowp[hi * 8 + e];
        a[8 + e] = rowp[16 + hi * 8 + e];
    }
    return a;
}

// B-matrix (32x16 bf16): lane n=lane&15 holds column n; K = hi*8+e and 16+hi*8+e (same as A).
__device__ __forceinline__ v16bf load_b_frag(const __bf16* colp, int hi) {
    v16bf b;
#pragma unroll
    for (int e = 0; e < 8; ++e) {
        b[e]     = colp[hi * 8 + e];
        b[8 + e] = colp[16 + hi * 8 + e];
    }
    return b;
}

__device__ __forceinline__ v8f wmma_bf16(v16bf a, v16bf b, v8f c) {
    return __builtin_amdgcn_wmma_f32_16x16x32_bf16(
        false, a, false, b, (short)0, c, false, false);
}

// ---------------- conversion kernels ----------------

__global__ void cvt_f32_bf16(const float* __restrict__ in, __bf16* __restrict__ out, int n) {
    int i = blockIdx.x * blockDim.x + threadIdx.x;
    if (i < n) out[i] = (__bf16)in[i];
}

__global__ void cvt_f32_bf16_t(const float* __restrict__ in, __bf16* __restrict__ out,
                               int rows, int cols) {
    int i = blockIdx.x * blockDim.x + threadIdx.x;
    if (i < rows * cols) {
        int r = i / cols, c = i - r * cols;
        out[(size_t)c * rows + r] = (__bf16)in[i];
    }
}

// ---------------- QKV projection: xb[4096][1024] @ Wt -> scatter Q,K,V ----------------
// grid(12, 64), block 256 (8 waves as 2x4). Block tile 64(M) x 256(N).
// A tile (64x32) staged in LDS via async copies, double buffered; B from global.
__global__ void qkv_gemm(const __bf16* __restrict__ xb,    // [B*N][D]
                         const __bf16* __restrict__ wt,    // [3D][D]  (W^T)
                         __bf16* __restrict__ qb,          // [B][H][N][Hd]
                         __bf16* __restrict__ kb,          // [B][H][N][Hd]
                         __bf16* __restrict__ vt)          // [B][H][Hd][N]
{
    __shared__ __bf16 abuf[2][64 * 32];

    const int tid  = threadIdx.x;
    const int lane = tid & 31;
    const int wave = tid >> 5;
    const int m  = lane & 15;
    const int hi = lane >> 4;
    const int blockrow0 = blockIdx.y * 64;
    const int row0 = blockrow0 + (wave >> 2) * 32;
    const int col0 = blockIdx.x * 256 + (wave & 3) * 64;
    const int wrow = (wave >> 2) * 32;

    // copy assignment: 256 threads x one 16B chunk = 4KB tile
    const int crow = tid >> 2;
    const int cch  = (tid & 3) * 8;
    const __bf16* asrc = xb + (size_t)(blockrow0 + crow) * EMBED + cch;

    async_load_b128(asrc, &abuf[0][crow * 32 + cch]);

    v8f acc[2][4] = {};
    int cur = 0;
    for (int k = 0; k < EMBED; k += 32) {
        wait_async0();
        __syncthreads();
        if (k + 32 < EMBED)
            async_load_b128(asrc + k + 32, &abuf[cur ^ 1][crow * 32 + cch]);

        v16bf a0 = load_a_frag(&abuf[cur][(wrow + m) * 32], hi);
        v16bf a1 = load_a_frag(&abuf[cur][(wrow + 16 + m) * 32], hi);
#pragma unroll
        for (int t = 0; t < 4; ++t) {
            v16bf b = load_b_frag(wt + (size_t)(col0 + t * 16 + m) * EMBED + k, hi);
            acc[0][t] = wmma_bf16(a0, b, acc[0][t]);
            acc[1][t] = wmma_bf16(a1, b, acc[1][t]);
        }
        cur ^= 1;
    }

    const int which = col0 >> 10;          // 0=q 1=k 2=v
    const int h     = (col0 & 1023) >> 6;
#pragma unroll
    for (int t = 0; t < 4; ++t) {
        const int hd = t * 16 + m;
#pragma unroll
        for (int rt = 0; rt < 2; ++rt) {
#pragma unroll
            for (int r = 0; r < 8; ++r) {
                const int row   = row0 + rt * 16 + r + 8 * hi;
                const int batch = row >> 11;
                const int n     = row & (NTOK - 1);
                const size_t bhh = (size_t)(batch * NHEAD + h);
                __bf16 val = (__bf16)acc[rt][t][r];
                if (which == 0)      qb[(bhh * NTOK + n) * HD + hd] = val;
                else if (which == 1) kb[(bhh * NTOK + n) * HD + hd] = val;
                else                 vt[(bhh * HD + hd) * NTOK + n] = val;
            }
        }
    }
}

// ---------------- attention: flash-style, 32 queries/wave, K/V staged in LDS ----------------
// grid(32 bh, 16), block 128 (4 waves share one bh): K/V tiles fetched once per block
__global__ void attn_kernel(const __bf16* __restrict__ qb,
                            const __bf16* __restrict__ kb,
                            const __bf16* __restrict__ vt,
                            __bf16* __restrict__ ctx)       // [B*N][D]
{
    __shared__ __bf16 kbufs[2][32 * 64];     // [key][hd]   4KB each
    __shared__ __bf16 vbufs[2][64 * 32];     // [hd][key]   4KB each
    __shared__ __bf16 pbuf[4][2][16 * 32];   // per-wave P tiles, 8KB

    const int tid  = threadIdx.x;
    const int lane = tid & 31;
    const int wave = tid >> 5;
    const int m  = lane & 15;
    const int hi = lane >> 4;
    const int bh = blockIdx.x;
    const int q0 = (blockIdx.y * 4 + wave) * 32;

    const __bf16* Q = qb + (size_t)bh * NTOK * HD;
    const __bf16* K = kb + (size_t)bh * NTOK * HD;
    const __bf16* V = vt + (size_t)bh * HD * NTOK;

    // Q fragments: 2 query tiles x 2 Hd chunks
    v16bf aq[2][2];
#pragma unroll
    for (int qt = 0; qt < 2; ++qt) {
        const __bf16* qrow = Q + (size_t)(q0 + qt * 16 + m) * HD;
        aq[qt][0] = load_a_frag(qrow, hi);
        aq[qt][1] = load_a_frag(qrow + 32, hi);
    }

    // K/V stage: 2 x 256 chunks of 16B over 128 threads (2 K + 2 V chunks per thread)
    const int kc0 = tid, kc1 = tid + 128;
    const int kkey0 = kc0 >> 3, kch0 = (kc0 & 7) * 8;
    const int kkey1 = kc1 >> 3, kch1 = (kc1 & 7) * 8;
    const int vhd0 = kc0 >> 2, vch0 = (kc0 & 3) * 8;
    const int vhd1 = kc1 >> 2, vch1 = (kc1 & 3) * 8;

    // prologue: stage j=0 into buffer 0
    async_load_b128(K + (size_t)kkey0 * HD + kch0, &kbufs[0][kkey0 * 64 + kch0]);
    async_load_b128(K + (size_t)kkey1 * HD + kch1, &kbufs[0][kkey1 * 64 + kch1]);
    async_load_b128(V + (size_t)vhd0 * NTOK + vch0, &vbufs[0][vhd0 * 32 + vch0]);
    async_load_b128(V + (size_t)vhd1 * NTOK + vch1, &vbufs[0][vhd1 * 32 + vch1]);

    float mrow[2][8], lrow[2][8];
    v8f o[2][4] = {};
#pragma unroll
    for (int qt = 0; qt < 2; ++qt)
#pragma unroll
        for (int r = 0; r < 8; ++r) { mrow[qt][r] = -3.0e38f; lrow[qt][r] = 0.f; }

    int cur = 0;
    for (int j = 0; j < NTOK; j += 32) {
        wait_async0();
        __syncthreads();
        if (j + 32 < NTOK) {
            const int nb = cur ^ 1;
            const size_t jn = j + 32;
            async_load_b128(K + (size_t)(jn + kkey0) * HD + kch0, &kbufs[nb][kkey0 * 64 + kch0]);
            async_load_b128(K + (size_t)(jn + kkey1) * HD + kch1, &kbufs[nb][kkey1 * 64 + kch1]);
            async_load_b128(V + (size_t)vhd0 * NTOK + jn + vch0, &vbufs[nb][vhd0 * 32 + vch0]);
            async_load_b128(V + (size_t)vhd1 * NTOK + jn + vch1, &vbufs[nb][vhd1 * 32 + vch1]);
        }

        // ---- K fragments from LDS (shared by all 4 waves) ----
        const __bf16* kl = kbufs[cur];
        v16bf bk00 = load_b_frag(kl + m * 64, hi);
        v16bf bk01 = load_b_frag(kl + m * 64 + 32, hi);
        v16bf bk10 = load_b_frag(kl + (16 + m) * 64, hi);
        v16bf bk11 = load_b_frag(kl + (16 + m) * 64 + 32, hi);

#pragma unroll
        for (int qt = 0; qt < 2; ++qt) {
            v8f s0 = {}, s1 = {};
            s0 = wmma_bf16(aq[qt][0], bk00, s0);
            s0 = wmma_bf16(aq[qt][1], bk01, s0);
            s1 = wmma_bf16(aq[qt][0], bk10, s1);
            s1 = wmma_bf16(aq[qt][1], bk11, s1);

            __bf16* P = pbuf[wave][qt];
#pragma unroll
            for (int r = 0; r < 8; ++r) {
                float v0 = s0[r] * ATT_SCALE;
                float v1 = s1[r] * ATT_SCALE;
                float tmax = fmaxf(v0, v1);
                tmax = fmaxf(tmax, __shfl_xor(tmax, 1, 32));
                tmax = fmaxf(tmax, __shfl_xor(tmax, 2, 32));
                tmax = fmaxf(tmax, __shfl_xor(tmax, 4, 32));
                tmax = fmaxf(tmax, __shfl_xor(tmax, 8, 32));
                float nm   = fmaxf(mrow[qt][r], tmax);
                float corr = __expf(mrow[qt][r] - nm);
                mrow[qt][r] = nm;
                lrow[qt][r] *= corr;
                o[qt][0][r] *= corr; o[qt][1][r] *= corr;
                o[qt][2][r] *= corr; o[qt][3][r] *= corr;

                float p0 = __expf(v0 - nm);
                float p1 = __expf(v1 - nm);
                float ps = p0 + p1;
                ps += __shfl_xor(ps, 1, 32);
                ps += __shfl_xor(ps, 2, 32);
                ps += __shfl_xor(ps, 4, 32);
                ps += __shfl_xor(ps, 8, 32);
                lrow[qt][r] += ps;
                P[(r + 8 * hi) * 32 + m]      = (__bf16)p0;
                P[(r + 8 * hi) * 32 + 16 + m] = (__bf16)p1;
            }
        }

        // ---- V fragments from LDS, reused by both q-tiles ----
        const __bf16* vl = vbufs[cur];
        v16bf bv0 = load_b_frag(vl + (0 * 16 + m) * 32, hi);
        v16bf bv1 = load_b_frag(vl + (1 * 16 + m) * 32, hi);
        v16bf bv2 = load_b_frag(vl + (2 * 16 + m) * 32, hi);
        v16bf bv3 = load_b_frag(vl + (3 * 16 + m) * 32, hi);

#pragma unroll
        for (int qt = 0; qt < 2; ++qt) {
            v16bf ap = load_a_frag(pbuf[wave][qt] + m * 32, hi);
            o[qt][0] = wmma_bf16(ap, bv0, o[qt][0]);
            o[qt][1] = wmma_bf16(ap, bv1, o[qt][1]);
            o[qt][2] = wmma_bf16(ap, bv2, o[qt][2]);
            o[qt][3] = wmma_bf16(ap, bv3, o[qt][3]);
        }
        cur ^= 1;
    }

    // ---- epilogue: normalize and store bf16 context [B*N][D] ----
    const int batch = bh >> 4;
    const int h     = bh & 15;
#pragma unroll
    for (int qt = 0; qt < 2; ++qt) {
#pragma unroll
        for (int r = 0; r < 8; ++r) {
            const float inv = 1.0f / lrow[qt][r];
            const size_t row = (size_t)(batch * NTOK + q0 + qt * 16 + r + 8 * hi);
            __bf16* dst = ctx + row * EMBED + h * HD + m;
            dst[0]  = (__bf16)(o[qt][0][r] * inv);
            dst[16] = (__bf16)(o[qt][1][r] * inv);
            dst[32] = (__bf16)(o[qt][2][r] * inv);
            dst[48] = (__bf16)(o[qt][3][r] * inv);
        }
    }
}

// ---------------- output projection: ctx[4096][1024] @ Wp^T + bias -> f32 ----------------
// grid(4, 64), block 256; A tile async-staged in LDS like qkv_gemm
__global__ void proj_gemm(const __bf16* __restrict__ ctx,
                          const __bf16* __restrict__ wpt,   // [D][D] (W^T)
                          const float* __restrict__ bias,
                          float* __restrict__ out)
{
    __shared__ __bf16 abuf[2][64 * 32];

    const int tid  = threadIdx.x;
    const int lane = tid & 31;
    const int wave = tid >> 5;
    const int m  = lane & 15;
    const int hi = lane >> 4;
    const int blockrow0 = blockIdx.y * 64;
    const int row0 = blockrow0 + (wave >> 2) * 32;
    const int col0 = blockIdx.x * 256 + (wave & 3) * 64;
    const int wrow = (wave >> 2) * 32;

    const int crow = tid >> 2;
    const int cch  = (tid & 3) * 8;
    const __bf16* asrc = ctx + (size_t)(blockrow0 + crow) * EMBED + cch;

    async_load_b128(asrc, &abuf[0][crow * 32 + cch]);

    v8f acc[2][4] = {};
    int cur = 0;
    for (int k = 0; k < EMBED; k += 32) {
        wait_async0();
        __syncthreads();
        if (k + 32 < EMBED)
            async_load_b128(asrc + k + 32, &abuf[cur ^ 1][crow * 32 + cch]);

        v16bf a0 = load_a_frag(&abuf[cur][(wrow + m) * 32], hi);
        v16bf a1 = load_a_frag(&abuf[cur][(wrow + 16 + m) * 32], hi);
#pragma unroll
        for (int t = 0; t < 4; ++t) {
            v16bf b = load_b_frag(wpt + (size_t)(col0 + t * 16 + m) * EMBED + k, hi);
            acc[0][t] = wmma_bf16(a0, b, acc[0][t]);
            acc[1][t] = wmma_bf16(a1, b, acc[1][t]);
        }
        cur ^= 1;
    }

#pragma unroll
    for (int t = 0; t < 4; ++t) {
        const int col = col0 + t * 16 + m;
        const float bv = bias[col];
#pragma unroll
        for (int rt = 0; rt < 2; ++rt) {
#pragma unroll
            for (int r = 0; r < 8; ++r) {
                const int row = row0 + rt * 16 + r + 8 * hi;
                out[(size_t)row * EMBED + col] = acc[rt][t][r] + bv;
            }
        }
    }
}

// ---------------- launch ----------------

extern "C" void kernel_launch(void* const* d_in, const int* in_sizes, int n_in,
                              void* d_out, int out_size, void* d_ws, size_t ws_size,
                              hipStream_t stream) {
    const float* x      = (const float*)d_in[0];
    const float* w_qkv  = (const float*)d_in[1];
    const float* w_proj = (const float*)d_in[2];
    const float* b_proj = (const float*)d_in[3];
    float* out = (float*)d_out;

    char* ws = (char*)d_ws;
    __bf16* xb     = (__bf16*)(ws);                          //  8 MB  [4096][1024]
    __bf16* wqkvt  = (__bf16*)(ws + ((size_t)8  << 20));     //  6 MB  [3072][1024]
    __bf16* wprojt = (__bf16*)(ws + ((size_t)14 << 20));     //  2 MB  [1024][1024]
    __bf16* qbuf   = (__bf16*)(ws + ((size_t)16 << 20));     //  8 MB  [B][H][N][Hd]
    __bf16* kbuf   = (__bf16*)(ws + ((size_t)24 << 20));     //  8 MB  [B][H][N][Hd]
    __bf16* vtb    = (__bf16*)(ws + ((size_t)32 << 20));     //  8 MB  [B][H][Hd][N]
    __bf16* ctx    = (__bf16*)(ws + ((size_t)40 << 20));     //  8 MB  [4096][1024]

    const int nx = NBATCH * NTOK * EMBED;                    // 4194304
    cvt_f32_bf16<<<nx / 256, 256, 0, stream>>>(x, xb, nx);
    cvt_f32_bf16_t<<<(EMBED * 3 * EMBED) / 256, 256, 0, stream>>>(w_qkv, wqkvt, EMBED, 3 * EMBED);
    cvt_f32_bf16_t<<<(EMBED * EMBED) / 256, 256, 0, stream>>>(w_proj, wprojt, EMBED, EMBED);

    qkv_gemm<<<dim3(12, 64), 256, 0, stream>>>(xb, wqkvt, qbuf, kbuf, vtb);
    attn_kernel<<<dim3(NBATCH * NHEAD, NTOK / 128), 128, 0, stream>>>(qbuf, kbuf, vtb, ctx);
    proj_gemm<<<dim3(4, 64), 256, 0, stream>>>(ctx, wprojt, b_proj, out);
}
